// alinet_highway_77163382440891
// MI455X (gfx1250) — matrix operand order repaired and checked by reading, not compiled, not run on hardware
//
#include <hip/hip_runtime.h>

typedef __attribute__((ext_vector_type(16))) _Float16 v16h;
typedef __attribute__((ext_vector_type(8)))  _Float16 v8h;
typedef __attribute__((ext_vector_type(8)))  float    v8f;
typedef __attribute__((ext_vector_type(4)))  int      v4i;

#define N_DIM 32768
#define D_DIM 2048
#define BN_EPS 1e-3f

#define BM 256
#define BN 128
#define BK 32
#define A_ELEMS (BM * BK)   // 8192 f16 = 16 KB per buffer
#define B_ELEMS (BN * BK)   // 4096 f16 =  8 KB per buffer

// -------- gfx1250 async global->LDS path (guarded; sync fallback) --------
#if defined(__has_builtin)
#  if __has_builtin(__builtin_amdgcn_global_load_async_to_lds_b128)
#    define HAVE_ASYNC_LDS 1
#  endif
#  if __has_builtin(__builtin_amdgcn_s_wait_asynccnt)
#    define WAIT_ASYNC(n) __builtin_amdgcn_s_wait_asynccnt(n)
#  endif
#endif
#ifndef HAVE_ASYNC_LDS
#  define HAVE_ASYNC_LDS 0
#endif
#ifndef WAIT_ASYNC
#  define WAIT_ASYNC(n) asm volatile("s_wait_asynccnt %0" ::"i"(n) : "memory")
#endif

#if HAVE_ASYNC_LDS
typedef __attribute__((address_space(1))) v4i gbl_v4i;
typedef __attribute__((address_space(3))) v4i lds_v4i;
#  define ASYNC_CP16(dst_lds, src_glb)                                        \
    __builtin_amdgcn_global_load_async_to_lds_b128(                           \
        (gbl_v4i*)(src_glb), (lds_v4i*)(dst_lds), 0, 0)
#endif

// ---------------- prep kernels ----------------

__global__ void bn_params_kernel(const float* __restrict__ gamma,
                                 const float* __restrict__ beta,
                                 const float* __restrict__ mean,
                                 const float* __restrict__ var,
                                 float* __restrict__ inv,
                                 float* __restrict__ bias) {
  int i = blockIdx.x * blockDim.x + threadIdx.x;
  if (i < D_DIM) {
    float iv = gamma[i] * rsqrtf(var[i] + BN_EPS);
    inv[i]  = iv;
    bias[i] = beta[i] - mean[i] * iv;
  }
}

// kernel[k][n] (f32, row-major) -> Bt[n][k] (f16), coalesced both sides
__global__ __launch_bounds__(256)
void transpose_f16_kernel(const float* __restrict__ x, _Float16* __restrict__ o) {
  __shared__ _Float16 tile[32][33];
  const int bx = blockIdx.x * 32;  // n base
  const int by = blockIdx.y * 32;  // k base
  const int tx = threadIdx.x & 31;
  const int ty = threadIdx.x >> 5;  // 0..7
#pragma unroll
  for (int j = 0; j < 32; j += 8)
    tile[ty + j][tx] = (_Float16)x[(size_t)(by + ty + j) * D_DIM + bx + tx];
  __syncthreads();
#pragma unroll
  for (int j = 0; j < 32; j += 8)
    o[(size_t)(bx + ty + j) * D_DIM + by + tx] = tile[tx][ty + j];
}

__global__ void bn_to_f16_kernel(const float* __restrict__ x,
                                 const float* __restrict__ inv,
                                 const float* __restrict__ bias,
                                 _Float16* __restrict__ o) {
  size_t i = ((size_t)blockIdx.x * blockDim.x + threadIdx.x) * 8;
  int col = (int)(i & (D_DIM - 1));  // 8 elems never cross a row boundary
#pragma unroll
  for (int j = 0; j < 8; ++j)
    o[i + j] = (_Float16)(x[i + j] * inv[col + j] + bias[col + j]);
}

// ---------------- fused WMMA GEMM + gated highway epilogue ----------------
// Block tile 256x128, 8 waves in 4x2 grid, 64x64 per wave (4x4 wmma accs).

__global__ __launch_bounds__(256)
void gated_gemm_kernel(const _Float16* __restrict__ Ah,   // [N,D] f16 = bn(x1)
                       const _Float16* __restrict__ Bt,   // [n][k] f16 = kernel^T
                       const float* __restrict__ inv,
                       const float* __restrict__ bias,
                       const float* __restrict__ x1,
                       const float* __restrict__ x2,
                       float* __restrict__ out) {
  __shared__ __align__(16) _Float16 sA[2][A_ELEMS];   // 32 KB
  __shared__ __align__(16) _Float16 sB[2][B_ELEMS];   // 16 KB

  const int t     = threadIdx.x;
  const int lane  = t & 31;
  const int wave  = t >> 5;     // 0..7
  const int half  = lane >> 4;  // 0/1
  const int r     = lane & 15;
  const int waveM = wave >> 1;  // 0..3 -> 64-row strip
  const int waveN = wave & 1;   // 0..1 -> 64-col strip
  const int blockM = blockIdx.y * BM;
  const int blockN = blockIdx.x * BN;

  // staging chunks of 8 f16 (16 B): A has 1024 chunks (4/thread), B 512 (2/thread)
  const int rowA0 = t >> 2,           offA0 = (t & 3) * 8;
  const int rowA1 = (t + 256) >> 2,   offA1 = ((t + 256) & 3) * 8;
  const int rowA2 = (t + 512) >> 2,   offA2 = ((t + 512) & 3) * 8;
  const int rowA3 = (t + 768) >> 2,   offA3 = ((t + 768) & 3) * 8;
  const int rowB0 = rowA0,            offB0 = offA0;
  const int rowB1 = rowA1,            offB1 = offA1;

  const _Float16* gA = Ah + (size_t)blockM * D_DIM;
  const _Float16* gB = Bt + (size_t)blockN * D_DIM;
  const int nK = D_DIM / BK;  // 64

  v8f acc[4][4] = {};
#if !HAVE_ASYNC_LDS
  v8h ra0, ra1, ra2, ra3, rb0, rb1;   // register double-buffer fallback
#endif

  auto issue_tile = [&](int kt) {  // start loading tile kt
    int k0 = kt * BK, buf = kt & 1;
    (void)buf;
#if HAVE_ASYNC_LDS
    ASYNC_CP16(&sA[buf][rowA0 * BK + offA0], gA + (size_t)rowA0 * D_DIM + k0 + offA0);
    ASYNC_CP16(&sA[buf][rowA1 * BK + offA1], gA + (size_t)rowA1 * D_DIM + k0 + offA1);
    ASYNC_CP16(&sA[buf][rowA2 * BK + offA2], gA + (size_t)rowA2 * D_DIM + k0 + offA2);
    ASYNC_CP16(&sA[buf][rowA3 * BK + offA3], gA + (size_t)rowA3 * D_DIM + k0 + offA3);
    ASYNC_CP16(&sB[buf][rowB0 * BK + offB0], gB + (size_t)rowB0 * D_DIM + k0 + offB0);
    ASYNC_CP16(&sB[buf][rowB1 * BK + offB1], gB + (size_t)rowB1 * D_DIM + k0 + offB1);
#else
    ra0 = *(const v8h*)(gA + (size_t)rowA0 * D_DIM + k0 + offA0);
    ra1 = *(const v8h*)(gA + (size_t)rowA1 * D_DIM + k0 + offA1);
    ra2 = *(const v8h*)(gA + (size_t)rowA2 * D_DIM + k0 + offA2);
    ra3 = *(const v8h*)(gA + (size_t)rowA3 * D_DIM + k0 + offA3);
    rb0 = *(const v8h*)(gB + (size_t)rowB0 * D_DIM + k0 + offB0);
    rb1 = *(const v8h*)(gB + (size_t)rowB1 * D_DIM + k0 + offB1);
#endif
  };
  auto commit_tile = [&](int kt) {  // sync path only: regs -> LDS
#if !HAVE_ASYNC_LDS
    int buf = kt & 1;
    *(v8h*)&sA[buf][rowA0 * BK + offA0] = ra0;
    *(v8h*)&sA[buf][rowA1 * BK + offA1] = ra1;
    *(v8h*)&sA[buf][rowA2 * BK + offA2] = ra2;
    *(v8h*)&sA[buf][rowA3 * BK + offA3] = ra3;
    *(v8h*)&sB[buf][rowB0 * BK + offB0] = rb0;
    *(v8h*)&sB[buf][rowB1 * BK + offB1] = rb1;
#else
    (void)kt;
#endif
  };

  auto compute = [&](int buf) {
    const _Float16* A = sA[buf];
    const _Float16* B = sB[buf];
    v16h bfrag[4];
#pragma unroll
    for (int sj = 0; sj < 4; ++sj) {             // B 32x16: lane r = col, K half*16..
      int col = waveN * 64 + sj * 16 + r;
      bfrag[sj] = *(const v16h*)&B[col * BK + half * 16];
    }
#pragma unroll
    for (int si = 0; si < 4; ++si) {             // A 16x32: lane r = row, K halves
      int arow = waveM * 64 + si * 16 + r;
      v8h lo = *(const v8h*)&A[arow * BK + half * 16];
      v8h hi = *(const v8h*)&A[arow * BK + half * 16 + 8];
      v16h afrag;
#pragma unroll
      for (int j = 0; j < 8; ++j) { afrag[j] = lo[j]; afrag[j + 8] = hi[j]; }
#pragma unroll
      for (int sj = 0; sj < 4; ++sj)
        acc[si][sj] = __builtin_amdgcn_wmma_f32_16x16x32_f16(
            false, afrag, false, bfrag[sj], (short)0, acc[si][sj], false, false);
    }
  };

  // prologue: tiles 0 and 1 in flight
  issue_tile(0);
  commit_tile(0);
  issue_tile(1);

  for (int k = 0; k < nK; ++k) {
#if HAVE_ASYNC_LDS
    if (k + 1 < nK) WAIT_ASYNC(6);   // tile k done; tile k+1 may be in flight
    else            WAIT_ASYNC(0);
#endif
    __syncthreads();                 // all waves' tile-k data visible
    compute(k & 1);
    __syncthreads();                 // everyone done reading buf (k&1)
    if (k + 1 < nK) commit_tile(k + 1);
    if (k + 2 < nK) issue_tile(k + 2);  // overwrites buf (k&1): safe post-barrier
  }

  // ---- fused epilogue: C/D layout -> VGPR i holds rows i (lanes 0-15) / i+8 ----
#pragma unroll
  for (int si = 0; si < 4; ++si) {
#pragma unroll
    for (int sj = 0; sj < 4; ++sj) {
      int gc = blockN + waveN * 64 + sj * 16 + r;
      float ic = inv[gc], bc = bias[gc];
      int gr0 = blockM + waveM * 64 + si * 16 + half * 8;
#pragma unroll
      for (int i = 0; i < 8; ++i) {
        size_t idx = (size_t)(gr0 + i) * D_DIM + gc;
        float h1 = x1[idx] * ic + bc;
        float h2 = x2[idx] * ic + bc;
        float g  = tanhf(acc[si][sj][i]);
        g = g > 0.0f ? g : 0.0f;
        out[idx] = tanhf(h2 + g * (h1 - h2));
      }
    }
  }
}

// ---------------- host launch ----------------

extern "C" void kernel_launch(void* const* d_in, const int* in_sizes, int n_in,
                              void* d_out, int out_size, void* d_ws, size_t ws_size,
                              hipStream_t stream) {
  const float* x1    = (const float*)d_in[0];
  const float* x2    = (const float*)d_in[1];
  const float* kern  = (const float*)d_in[2];
  const float* gamma = (const float*)d_in[3];
  const float* beta  = (const float*)d_in[4];
  const float* mmean = (const float*)d_in[5];
  const float* mvar  = (const float*)d_in[6];
  float* out = (float*)d_out;

  char* w = (char*)d_ws;
  _Float16* Ah = (_Float16*)w;                                 // 128 MB
  _Float16* Bt = (_Float16*)(w + (size_t)N_DIM * D_DIM * 2);   //   8 MB
  float* inv   = (float*)(w + (size_t)N_DIM * D_DIM * 2
                            + (size_t)D_DIM * D_DIM * 2);
  float* bias  = inv + D_DIM;

  bn_params_kernel<<<D_DIM / 256, 256, 0, stream>>>(gamma, beta, mmean, mvar, inv, bias);
  dim3 tg(D_DIM / 32, D_DIM / 32);
  transpose_f16_kernel<<<tg, 256, 0, stream>>>(kern, Bt);
  bn_to_f16_kernel<<<(unsigned)((size_t)N_DIM * D_DIM / (256 * 8)), 256, 0, stream>>>(x1, inv, bias, Ah);

  dim3 grid(D_DIM / BN, N_DIM / BM);
  gated_gemm_kernel<<<grid, 256, 0, stream>>>(Ah, Bt, inv, bias, x1, x2, out);
}
